// Attention_45818711114233
// MI455X (gfx1250) — compile-verified, matrix operand order checked by reading, not run on hardware
//
#include <hip/hip_runtime.h>
#include <stdint.h>
#include <math.h>

// ---------------------------------------------------------------- constants
#define S_    2048
#define H_    2048
#define NH_   32
#define NKV_  8
#define HD_   64
#define NKVD_ 512          // NKV*HD
#define SCALE_ 0.125f      // 64^-0.5

typedef __attribute__((ext_vector_type(16))) __bf16 v16bf;
typedef __attribute__((ext_vector_type(8)))  __bf16 v8bf;
typedef __attribute__((ext_vector_type(8)))  float  v8f;

// ---------------------------------------------------------------- helpers
__device__ __forceinline__ __bf16 f2bf(float f) {
  union { float f; uint32_t u; } v; v.f = f;
  uint32_t r = (v.u + 0x7FFFu + ((v.u >> 16) & 1u)) >> 16;
  union { uint16_t s; __bf16 b; } o; o.s = (uint16_t)r; return o.b;
}
__device__ __forceinline__ float bf2f(__bf16 b) {
  union { uint16_t s; __bf16 b; } i; i.b = b;
  union { uint32_t u; float f; } o; o.u = ((uint32_t)i.s) << 16; return o.f;
}
__device__ __forceinline__ v8f vzero8() {
  v8f z;
#pragma unroll
  for (int i = 0; i < 8; ++i) z[i] = 0.f;
  return z;
}
// 16-bit A/B fragment per CDNA5 ISA layout: lane holds one row, with two
// contiguous 8-element runs at columns kb0 and 16+kb0 (kb0 = (lane>=16)*8).
__device__ __forceinline__ v16bf load_frag(const __bf16* rowptr, int kb0) {
  v8bf lo = *(const v8bf*)(rowptr + kb0);
  v8bf hi = *(const v8bf*)(rowptr + 16 + kb0);
  return __builtin_shufflevector(lo, hi, 0,1,2,3,4,5,6,7,8,9,10,11,12,13,14,15);
}
__device__ __forceinline__ v8f wmma_bf16(v16bf a, v16bf b, v8f c) {
  return __builtin_amdgcn_wmma_f32_16x16x32_bf16(false, a, false, b,
                                                 (short)0, c, false, false);
}
// CDNA5 async DMA: global (16B per lane) -> LDS, tracked by ASYNCcnt.
__device__ __forceinline__ void async_copy_b128(const __bf16* g, const __bf16* l) {
  uint64_t ga = (uint64_t)(uintptr_t)g;
  uint32_t la = (uint32_t)(uintptr_t)l;   // low 32 bits = LDS byte offset
  asm volatile("global_load_async_to_lds_b128 %0, %1, off"
               :: "v"(la), "v"(ga) : "memory");
}

// ---------------------------------------------------------------- converts
__global__ __launch_bounds__(256) void cvt_kernel(const float* __restrict__ in,
                                                  __bf16* __restrict__ out, int n) {
  int i = blockIdx.x * 256 + threadIdx.x;
  if (i < n) out[i] = f2bf(in[i]);
}
// w[K][N] f32  ->  wT[N][K] bf16 (so B fragments are contiguous along K)
__global__ __launch_bounds__(256) void cvtT_kernel(const float* __restrict__ in,
                                                   __bf16* __restrict__ outT,
                                                   int K, int N) {
  int i = blockIdx.x * 256 + threadIdx.x;
  if (i >= K * N) return;
  int k = i / N, n = i - k * N;
  outT[(size_t)n * K + k] = f2bf(in[i]);
}

// ---------------------------------------------------------------- GEMM
// C[M][N] = A[M][K] (bf16 row-major) x B (given as BT[N][K] bf16)
// block = 256 thr = 8 waves (2x4); wave tile 32x64; block tile 64x256.
// Double-buffered LDS staging via global_load_async_to_lds_b128 (ASYNCcnt):
// each thread issues 5 b128 copies per 32-wide k-tile (1 A + 4 B).
template <int OUT_F32>
__global__ __launch_bounds__(256)
void gemm_bf16_kernel(const __bf16* __restrict__ A, const __bf16* __restrict__ BT,
                      void* __restrict__ Cout, int M, int N, int K) {
  __shared__ __bf16 lsA[2][64][32];    //  2 x 4 KB
  __shared__ __bf16 lsB[2][256][32];   //  2 x 16 KB

  const int lane = threadIdx.x & 31;
  const int wid  = threadIdx.x >> 5;
  const int half = lane >> 4;
  const int lm   = lane & 15;
  const int kb0  = half * 8;
  const int mw   = (wid >> 2) * 32;             // wave row inside block tile
  const int nw   = (wid & 3) * 64;              // wave col inside block tile
  const int m0blk = blockIdx.y * 64;
  const int n0blk = blockIdx.x * 256;

  const int crow = threadIdx.x >> 2;            // 0..63 copy row
  const int cch  = (threadIdx.x & 3) * 8;       // 16B chunk within k-tile

  v8f acc[2][4];
#pragma unroll
  for (int mt = 0; mt < 2; ++mt)
#pragma unroll
    for (int nt = 0; nt < 4; ++nt) acc[mt][nt] = vzero8();

  auto issue_tile = [&](int buf, int k0) {
    async_copy_b128(A + (size_t)(m0blk + crow) * K + k0 + cch,
                    &lsA[buf][crow][cch]);
#pragma unroll
    for (int i = 0; i < 4; ++i)
      async_copy_b128(BT + (size_t)(n0blk + crow + i * 64) * K + k0 + cch,
                      &lsB[buf][crow + i * 64][cch]);
  };

  const int steps = K >> 5;
  issue_tile(0, 0);
  for (int kt = 0; kt < steps; ++kt) {
    if (kt + 1 < steps) {
      issue_tile((kt + 1) & 1, (kt + 1) << 5);
      asm volatile("s_wait_asynccnt 0x5" ::: "memory");  // current tile resident
    } else {
      asm volatile("s_wait_asynccnt 0x0" ::: "memory");
    }
    __syncthreads();
    const __bf16* bufA = &lsA[kt & 1][0][0];
    const __bf16* bufB = &lsB[kt & 1][0][0];
    v16bf am[2], bn[4];
    am[0] = load_frag(bufA + (mw + lm) * 32, kb0);
    am[1] = load_frag(bufA + (mw + 16 + lm) * 32, kb0);
#pragma unroll
    for (int nt = 0; nt < 4; ++nt)
      bn[nt] = load_frag(bufB + (nw + nt * 16 + lm) * 32, kb0);
#pragma unroll
    for (int mt = 0; mt < 2; ++mt)
#pragma unroll
      for (int nt = 0; nt < 4; ++nt)
        acc[mt][nt] = wmma_bf16(am[mt], bn[nt], acc[mt][nt]);
    __syncthreads();   // protect buffer from next iteration's DMA
  }
  // C/D layout: element r -> row (r + 8*half), col = lane&15
#pragma unroll
  for (int mt = 0; mt < 2; ++mt)
#pragma unroll
    for (int nt = 0; nt < 4; ++nt)
#pragma unroll
      for (int r = 0; r < 8; ++r) {
        int row = m0blk + mw + mt * 16 + r + half * 8;
        int col = n0blk + nw + nt * 16 + lm;
        if (OUT_F32) ((float*)Cout)[(size_t)row * N + col] = acc[mt][nt][r];
        else ((__bf16*)Cout)[(size_t)row * N + col] = f2bf(acc[mt][nt][r]);
      }
}

// ---------------------------------------------------------------- RoPE (+scale, +V transpose)
__global__ __launch_bounds__(256)
void rope_kernel(const __bf16* __restrict__ qtmp, const __bf16* __restrict__ ktmp,
                 const __bf16* __restrict__ vtmp, const float* __restrict__ cosp,
                 const float* __restrict__ sinp, __bf16* __restrict__ qb,
                 __bf16* __restrict__ kb, __bf16* __restrict__ vT) {
  int idx = blockIdx.x * 256 + threadIdx.x;
  if (idx >= S_ * H_) return;
  int s = idx >> 11, c = idx & (H_ - 1);
  int d = c & (HD_ - 1);
  float cs = cosp[s * HD_ + d], sn = sinp[s * HD_ + d];
  {
    float v = bf2f(qtmp[idx]);
    float p = (d < 32) ? -bf2f(qtmp[idx + 32]) : bf2f(qtmp[idx - 32]);
    qb[idx] = f2bf(SCALE_ * (v * cs + p * sn));   // fold softmax scale into Q
  }
  if (c < NKVD_) {
    int ki = s * NKVD_ + c;
    float v = bf2f(ktmp[ki]);
    float p = (d < 32) ? -bf2f(ktmp[ki + 32]) : bf2f(ktmp[ki - 32]);
    kb[ki] = f2bf(v * cs + p * sn);
    vT[(size_t)c * S_ + s] = vtmp[ki];            // Vt[d_global][s]
  }
}

// ---------------------------------------------------------------- attention
// One wave per (head, 16-query tile). Online softmax; causal handled
// analytically with exact per-wave trip counts. block = 128 thr = 4 waves.
__global__ __launch_bounds__(128)
void attn_kernel(const __bf16* __restrict__ qb,   // [S][2048] (scaled, roped)
                 const __bf16* __restrict__ kb,   // [S][512]  (roped)
                 const __bf16* __restrict__ vT,   // [512][S]
                 __bf16* __restrict__ aout) {     // [S][2048]
  __shared__ __bf16 pls[4][16][32];               // per-wave P relayout buffer
  const int lane = threadIdx.x & 31;
  const int wid  = threadIdx.x >> 5;
  const int half = lane >> 4;
  const int lm   = lane & 15;
  const int kb0  = half * 8;
  const int w    = blockIdx.x * 4 + wid;
  const int h    = w >> 7;                        // 128 q-tiles per head
  const int qg   = (w & 127) << 4;
  const int kvh  = h >> 2;                        // GQA group of 4

  const __bf16* qrow = qb + (size_t)(qg + lm) * H_ + h * HD_;
  v16bf aq0 = load_frag(qrow +  0, kb0);
  v16bf aq1 = load_frag(qrow + 32, kb0);

  v8f o[4];
#pragma unroll
  for (int c = 0; c < 4; ++c) o[c] = vzero8();
  float mrow[8], lrow[8];
#pragma unroll
  for (int r = 0; r < 8; ++r) { mrow[r] = -INFINITY; lrow[r] = 0.f; }

  const int nkt = (qg + 47) >> 5;                 // ceil((qg+16)/32) key tiles
  for (int kt = 0; kt < nkt; ++kt) {
    const int ks = kt << 5;
    if (kt + 1 < nkt) {                           // -> global_prefetch_b8
      __builtin_prefetch(kb + (size_t)(ks + 32 + lm) * NKVD_ + kvh * HD_, 0, 1);
      __builtin_prefetch(kb + (size_t)(ks + 48 + lm) * NKVD_ + kvh * HD_, 0, 1);
    }
    // ---- issue ALL independent global loads first (K frags + V frags) so
    //      their latency overlaps the score WMMAs and the softmax VALU work.
    v16bf bk[2][2], bv[4];
#pragma unroll
    for (int st = 0; st < 2; ++st) {
      const __bf16* krow = kb + (size_t)(ks + st * 16 + lm) * NKVD_ + kvh * HD_;
      bk[st][0] = load_frag(krow +  0, kb0);
      bk[st][1] = load_frag(krow + 32, kb0);
    }
#pragma unroll
    for (int c = 0; c < 4; ++c)
      bv[c] = load_frag(vT + (size_t)(kvh * HD_ + c * 16 + lm) * S_ + ks, kb0);
    // ---- scores: S[16q x 32k] = Q(16xHD) . K^T, 4 WMMAs
    v8f sc[2];
#pragma unroll
    for (int st = 0; st < 2; ++st) {
      v8f z = vzero8();
      z = wmma_bf16(aq0, bk[st][0], z);
      sc[st] = wmma_bf16(aq1, bk[st][1], z);
    }
    // ---- causal mask (row = qg + r + 8*half, col = ks + st*16 + lane&15)
#pragma unroll
    for (int r = 0; r < 8; ++r) {
      int qrw = qg + r + half * 8;
      if (ks + lm      > qrw) sc[0][r] = -INFINITY;
      if (ks + 16 + lm > qrw) sc[1][r] = -INFINITY;
    }
    // ---- row max across the 16 lanes of each half-wave
    float tmax[8];
#pragma unroll
    for (int r = 0; r < 8; ++r) tmax[r] = fmaxf(sc[0][r], sc[1][r]);
#pragma unroll
    for (int off = 1; off < 16; off <<= 1)
#pragma unroll
      for (int r = 0; r < 8; ++r)
        tmax[r] = fmaxf(tmax[r], __shfl_xor(tmax[r], off, 32));
    // ---- online softmax update
    float p0[8], p1[8], rsum[8];
#pragma unroll
    for (int r = 0; r < 8; ++r) {
      float nm = fmaxf(mrow[r], tmax[r]);
      float resc = __expf(mrow[r] - nm);
      mrow[r] = nm;
      p0[r] = __expf(sc[0][r] - nm);
      p1[r] = __expf(sc[1][r] - nm);
      rsum[r] = p0[r] + p1[r];
      lrow[r] *= resc;
#pragma unroll
      for (int c = 0; c < 4; ++c) o[c][r] *= resc;
    }
#pragma unroll
    for (int off = 1; off < 16; off <<= 1)
#pragma unroll
      for (int r = 0; r < 8; ++r) rsum[r] += __shfl_xor(rsum[r], off, 32);
#pragma unroll
    for (int r = 0; r < 8; ++r) lrow[r] += rsum[r];
    // ---- relayout P: C-layout f32 -> A-layout bf16 via per-wave LDS
#pragma unroll
    for (int r = 0; r < 8; ++r) {
      pls[wid][r + half * 8][lm]      = f2bf(p0[r]);
      pls[wid][r + half * 8][16 + lm] = f2bf(p1[r]);
    }
    asm volatile("s_wait_dscnt 0x0" ::: "memory");
    v16bf ap = load_frag(&pls[wid][lm][0], kb0);
    // ---- O += P(16x32) . V(32xHD), 4 WMMAs (V frags already resident)
#pragma unroll
    for (int c = 0; c < 4; ++c)
      o[c] = wmma_bf16(ap, bv[c], o[c]);
  }
  // ---- epilogue: normalize and scatter to [S][NH*HD]
  float inv[8];
#pragma unroll
  for (int r = 0; r < 8; ++r) inv[r] = 1.f / lrow[r];
#pragma unroll
  for (int c = 0; c < 4; ++c)
#pragma unroll
    for (int r = 0; r < 8; ++r) {
      int row = qg + r + half * 8;
      int col = h * HD_ + c * 16 + lm;
      aout[(size_t)row * H_ + col] = f2bf(o[c][r] * inv[r]);
    }
}

// ---------------------------------------------------------------- launch
extern "C" void kernel_launch(void* const* d_in, const int* in_sizes, int n_in,
                              void* d_out, int out_size, void* d_ws, size_t ws_size,
                              hipStream_t stream) {
  const float* x   = (const float*)d_in[0];
  const float* wq  = (const float*)d_in[1];
  const float* wk  = (const float*)d_in[2];
  const float* wv  = (const float*)d_in[3];
  const float* wo  = (const float*)d_in[4];
  const float* cs  = (const float*)d_in[5];
  const float* sn  = (const float*)d_in[6];
  // d_in[7] = causal mask: handled analytically, not read.

  char* ws = (char*)d_ws;
  const size_t MB = 1ull << 20;
  __bf16* xb   = (__bf16*)(ws + 0 * MB);   // 8 MB  (reused as aout)
  __bf16* wqT  = (__bf16*)(ws + 8 * MB);   // 8 MB  (reused as qb)
  __bf16* wkT  = (__bf16*)(ws + 16 * MB);  // 2 MB  (reused as kb)
  __bf16* wvT  = (__bf16*)(ws + 18 * MB);  // 2 MB  (reused as vT)
  __bf16* woT  = (__bf16*)(ws + 20 * MB);  // 8 MB
  __bf16* qtmp = (__bf16*)(ws + 28 * MB);  // 8 MB
  __bf16* ktmp = (__bf16*)(ws + 36 * MB);  // 2 MB
  __bf16* vtmp = (__bf16*)(ws + 38 * MB);  // 2 MB  (total 40 MB)
  __bf16* qb = wqT;  __bf16* kb = wkT;  __bf16* vT = wvT;  __bf16* aout = xb;

  // 1) precision conversion + weight transposes (bf16)
  cvt_kernel <<<(S_ * H_ + 255) / 256, 256, 0, stream>>>(x, xb, S_ * H_);
  cvtT_kernel<<<(H_ * H_    + 255) / 256, 256, 0, stream>>>(wq, wqT, H_, H_);
  cvtT_kernel<<<(H_ * NKVD_ + 255) / 256, 256, 0, stream>>>(wk, wkT, H_, NKVD_);
  cvtT_kernel<<<(H_ * NKVD_ + 255) / 256, 256, 0, stream>>>(wv, wvT, H_, NKVD_);
  cvtT_kernel<<<(H_ * H_    + 255) / 256, 256, 0, stream>>>(wo, woT, H_, H_);

  // 2) QKV projections (bf16 WMMA, f32 accum, bf16 out, async-LDS staged)
  gemm_bf16_kernel<0><<<dim3(H_ / 256,    S_ / 64), 256, 0, stream>>>(xb, wqT, qtmp, S_, H_,    H_);
  gemm_bf16_kernel<0><<<dim3(NKVD_ / 256, S_ / 64), 256, 0, stream>>>(xb, wkT, ktmp, S_, NKVD_, H_);
  gemm_bf16_kernel<0><<<dim3(NKVD_ / 256, S_ / 64), 256, 0, stream>>>(xb, wvT, vtmp, S_, NKVD_, H_);

  // 3) RoPE (+ fold softmax scale into Q) and V transpose
  rope_kernel<<<(S_ * H_ + 255) / 256, 256, 0, stream>>>(qtmp, ktmp, vtmp, cs, sn, qb, kb, vT);

  // 4) causal GQA flash attention (one wave per head x 16 queries)
  attn_kernel<<<(NH_ * (S_ / 16)) / 4, 128, 0, stream>>>(qb, kb, vT, aout);

  // 5) output projection -> f32
  gemm_bf16_kernel<1><<<dim3(H_ / 256, S_ / 64), 256, 0, stream>>>(aout, woT, d_out, S_, H_, H_);
}
